// NBMBlock_76587856823102
// MI455X (gfx1250) — matrix operand order repaired and checked by reading, not compiled, not run on hardware
//
#include <hip/hip_runtime.h>

// Problem constants (match reference)
#define B_DIM 128
#define F_DIM 128
#define D_DIM 512
#define K_DIM 32
#define H1    64
#define H2    32

#define SAMPLES_PER_BLOCK 512     // one full batch row b (all of D)
#define TILES_PER_WAVE 4          // 4 x 16-row M-tiles = 64 samples per wave

typedef __attribute__((ext_vector_type(16))) __bf16 v16bf;
typedef __attribute__((ext_vector_type(8)))  float  v8f;

// Fused gather + 3-layer expert MLP, B-fragments reused across 4 M-tiles.
// grid.x = B * K (128*32 = 4096) ; block = 256 (8 wave32)
// block (b, k): all 512 samples of batch row b through expert k.
// wave w: samples [w*64, w*64+64) as 4 WMMA M-tiles.
__global__ __launch_bounds__(256)
void nbm_fused_wmma(const float* __restrict__ net,
                    const float* __restrict__ w1,
                    const float* __restrict__ b1,
                    const float* __restrict__ w2,
                    const float* __restrict__ b2,
                    const float* __restrict__ w3,
                    const float* __restrict__ b3,
                    const int*   __restrict__ indices,
                    float*       __restrict__ out)
{
    __shared__ float w1S[H1 * 2];     // w1[k,o,c] -> [o*2+c]
    __shared__ float b1S[H1];
    __shared__ float w2S[H2 * H1];    // w2[k,o,c] -> [o*H1+c]
    __shared__ float b2S[H2];
    __shared__ float w3S[H2];
    __shared__ float gS[SAMPLES_PER_BLOCK * 2];   // gathered inputs [d*2+c]

    const int tid   = threadIdx.x;
    const int k     = blockIdx.x % K_DIM;
    const int b     = blockIdx.x / K_DIM;   // 0..127
    const int sBase = b * D_DIM;            // linear sample base (= b*D)

    // ---- cooperative stage of expert-k weights into LDS (once per block) ----
    if (tid < H1 * 2) w1S[tid] = w1[k * H1 * 2 + tid];
    if (tid < H1)     b1S[tid] = b1[k * H1 + tid];
#pragma unroll
    for (int i = 0; i < (H2 * H1) / 256; ++i)
        w2S[i * 256 + tid] = w2[k * H2 * H1 + i * 256 + tid];
    if (tid < H2) { b2S[tid] = b2[k * H2 + tid]; w3S[tid] = w3[k * H2 + tid]; }

    // ---- gather g = net[b, indices[d, c]] for 512 samples x ORDER=2 ----
#pragma unroll
    for (int i = 0; i < (SAMPLES_PER_BLOCK * 2) / 256; ++i) {
        const int e = i * 256 + tid;
        const int d = e >> 1;
        const int c = e & 1;
        gS[e] = net[b * F_DIM + indices[d * 2 + c]];
    }
    __syncthreads();

    const int lane = tid & 31;
    const int wave = tid >> 5;
    const int hi   = lane >> 4;   // half-wave group
    const int n    = lane & 15;   // A: row within tile / B,C: column
    const int wBase = wave * (TILES_PER_WAVE * 16);   // wave's local sample base

    // ---- B fragments from w2, built ONCE per wave, reused for 4 M-tiles ----
    // B 32x16 bf16 layout: lane(col n), VGPR j holds K = hi*16 + 2j (+1)
    v16bf B00, B01, B10, B11;     // B{Ntile}{Kstep}
#pragma unroll
    for (int j = 0; j < 8; ++j) {
        const int kk = (hi << 4) + (j << 1);
#pragma unroll
        for (int e = 0; e < 2; ++e) {
            B00[2 * j + e] = (__bf16)w2S[(n)      * H1 + kk + e];
            B01[2 * j + e] = (__bf16)w2S[(n)      * H1 + 32 + kk + e];
            B10[2 * j + e] = (__bf16)w2S[(16 + n) * H1 + kk + e];
            B11[2 * j + e] = (__bf16)w2S[(16 + n) * H1 + 32 + kk + e];
        }
    }

    const float bc0 = b2S[n], bc1 = b2S[16 + n];
    const float w3a = w3S[n];
    const float w3b = w3S[16 + n];
    const float b3v = b3[k];

#pragma unroll
    for (int ti = 0; ti < TILES_PER_WAVE; ++ti) {
        const int m0 = wBase + ti * 16;

        // ---- layer 1 (VALU, fp32) -> A fragments (bf16) ----
        // A 16x32 bf16 layout: lane(row m = n), VGPR j holds
        //   K = hi*16 + 2j (+1)
        const float g0 = gS[(m0 + n) * 2 + 0];
        const float g1 = gS[(m0 + n) * 2 + 1];

        v16bf A0, A1;             // K-steps: H1 channels [0,32) and [32,64)
#pragma unroll
        for (int j = 0; j < 8; ++j) {
            const int kb = (hi << 4) + (j << 1);
#pragma unroll
            for (int e = 0; e < 2; ++e) {
                const int c0 = kb + e;
                const float x0 = fmaxf(g0 * w1S[c0 * 2] + g1 * w1S[c0 * 2 + 1] + b1S[c0], 0.0f);
                A0[2 * j + e] = (__bf16)x0;
                const int c1 = 32 + c0;
                const float x1 = fmaxf(g0 * w1S[c1 * 2] + g1 * w1S[c1 * 2 + 1] + b1S[c1], 0.0f);
                A1[2 * j + e] = (__bf16)x1;
            }
        }

        // ---- layer 2: C = A*B + b2, f32 accumulate on the matrix pipe ----
        // C 16x16 f32 layout: VGPR r, row m = r + 8*hi, col = n
        v8f C0, C1;
#pragma unroll
        for (int r = 0; r < 8; ++r) { C0[r] = bc0; C1[r] = bc1; }

        C0 = __builtin_amdgcn_wmma_f32_16x16x32_bf16(false, A0, false, B00, (short)0, C0, false, false);
        C0 = __builtin_amdgcn_wmma_f32_16x16x32_bf16(false, A1, false, B01, (short)0, C0, false, false);
        C1 = __builtin_amdgcn_wmma_f32_16x16x32_bf16(false, A0, false, B10, (short)0, C1, false, false);
        C1 = __builtin_amdgcn_wmma_f32_16x16x32_bf16(false, A1, false, B11, (short)0, C1, false, false);

        // ---- layer 3: relu, scale by w3 per column, butterfly-reduce columns ----
        float acc[8];
#pragma unroll
        for (int r = 0; r < 8; ++r)
            acc[r] = fmaxf(C0[r], 0.0f) * w3a + fmaxf(C1[r], 0.0f) * w3b;

#pragma unroll
        for (int r = 0; r < 8; ++r) {
            acc[r] += __shfl_xor(acc[r], 1, 32);
            acc[r] += __shfl_xor(acc[r], 2, 32);
            acc[r] += __shfl_xor(acc[r], 4, 32);
            acc[r] += __shfl_xor(acc[r], 8, 32);
        }

        if (n == 0) {             // lanes 0 and 16 hold rows r and r+8
#pragma unroll
            for (int r = 0; r < 8; ++r) {
                const int srow = m0 + r + (hi << 3);             // local sample
                out[(size_t)(sBase + srow) * K_DIM + k] = acc[r] + b3v;
            }
        }
    }
}

extern "C" void kernel_launch(void* const* d_in, const int* in_sizes, int n_in,
                              void* d_out, int out_size, void* d_ws, size_t ws_size,
                              hipStream_t stream) {
    (void)in_sizes; (void)n_in; (void)out_size; (void)d_ws; (void)ws_size;
    const float* net     = (const float*)d_in[0];
    const float* w1      = (const float*)d_in[1];
    const float* b1      = (const float*)d_in[2];
    const float* w2      = (const float*)d_in[3];
    const float* b2      = (const float*)d_in[4];
    const float* w3      = (const float*)d_in[5];
    const float* b3      = (const float*)d_in[6];
    const int*   indices = (const int*)d_in[7];
    float* out = (float*)d_out;

    dim3 grid(B_DIM * K_DIM);   // 128 batch rows * 32 experts = 4096 blocks
    dim3 block(256);            // 8 wave32, each wave: 64 samples (4 M-tiles)
    hipLaunchKernelGGL(nbm_fused_wmma, grid, block, 0, stream,
                       net, w1, b1, w2, b2, w3, b3, indices, out);
}
